// top_GRU_60189671686548
// MI455X (gfx1250) — compile-verified, matrix operand order checked by reading, not run on hardware
//
#include <hip/hip_runtime.h>

// ---------------------------------------------------------------------------
// GRU on MI455X (gfx1250, wave32, WMMA).
//   Phase 1: xg = x @ W_i + bias  via V_WMMA_F32_16X16X4_F32 (exact fp32),
//            written straight into the r/z/n slots of states_seq in d_out.
//   Phase 2: sequential scan, W_h resident in VGPRs as bf16 (192 VGPRs/wave),
//            24 v_wmma_f32_16x16x32_bf16 per wave per step, grid barrier.
//   Phase 3: final FC fused at the end of phase 2 (fp32 WMMA).
// ---------------------------------------------------------------------------

typedef __attribute__((ext_vector_type(2)))  float  v2f;
typedef __attribute__((ext_vector_type(8)))  float  v8f;
typedef __attribute__((ext_vector_type(16))) __bf16 v16bf;

#define B_  32
#define T_  2048
#define H_  256
#define G_  768              // 3*H
#define OUTBASE 8192         // B_*H_ floats of `output` precede states_seq

union BfVec { v16bf v; unsigned short u[16]; };

__device__ __forceinline__ unsigned short f2bf(float f) {
  unsigned int u = __float_as_uint(f);
  u += 0x7FFFu + ((u >> 16) & 1u);        // round-to-nearest-even
  return (unsigned short)(u >> 16);
}

__device__ __forceinline__ float sigmoidf_(float x) {
  return 1.0f / (1.0f + __expf(-x));
}

// ---------------------------------------------------------------------------
// Phase 1: xg(b,t,g) = x(b,t,:) @ W_i(:,g) + bias(g)
// Stored at states_seq slot s = 1 + g/256, column j = g%256, time t.
// One wave = 8 consecutive 16-row M-tiles of one 16-col N-tile.
// ---------------------------------------------------------------------------
__global__ __launch_bounds__(256) void xg_gemm_kernel(
    const float* __restrict__ x, const float* __restrict__ Wi,
    const float* __restrict__ bias, float* __restrict__ out)
{
  const int lane  = threadIdx.x & 31;
  const int wid   = (blockIdx.x * blockDim.x + threadIdx.x) >> 5;  // 0..24575
  const int ntile = wid % 48;
  const int mgrp  = wid / 48;                                      // 0..511
  const int n     = lane & 15;
  const int kg    = lane >> 4;
  const int col   = ntile * 16 + n;                                // 0..767

  // Resident B fragments: 64 K-chunks of 4 (fp32 WMMA).
  // B(k,c): lane = n + 16*((k&3)>>1), vgpr = k&1.
  v2f bfr[64];
#pragma unroll
  for (int c = 0; c < 64; ++c) {
    const int k0 = c * 4 + 2 * kg;
    bfr[c].x = Wi[(size_t)k0 * G_ + col];
    bfr[c].y = Wi[(size_t)(k0 + 1) * G_ + col];
  }
  const float bcol = bias[col];
  const int s = 1 + (col >> 8);          // gate slot in states_seq (1..3)
  const int j = col & 255;

  for (int mi = 0; mi < 8; ++mi) {
    const int mtile = mgrp * 8 + mi;
    const float* xr = x + (size_t)(mtile * 16 + n) * H_;   // A row (m = lane&15)
    v8f acc = {};
#pragma unroll
    for (int c = 0; c < 64; ++c) {
      const int k0 = c * 4 + 2 * kg;
      v2f a; a.x = xr[k0]; a.y = xr[k0 + 1];
      acc = __builtin_amdgcn_wmma_f32_16x16x4_f32(
          false, a, false, bfr[c], (short)0, acc, false, false);
    }
    // C layout: n = lane&15 (already col), m = v + 8*kg
    const int rowBase = mtile * 16 + 8 * kg;
#pragma unroll
    for (int v = 0; v < 8; ++v) {
      const int r = rowBase + v;                 // 0..65535
      const int b = r >> 11, t = r & 2047;
      const size_t off =
          OUTBASE + ((((size_t)b * 4 + s) * T_ + t) * H_) + j;
      out[off] = acc[v] + bcol;
    }
  }
}

// ---------------------------------------------------------------------------
// Phase 2: sequential scan. 32 wave-jobs = 2 M-tiles x 16 column-tiles.
// 4 blocks x 256 threads; grid barrier each step via monotonic counter.
// ---------------------------------------------------------------------------
__global__ __launch_bounds__(256) void gru_scan_kernel(
    const float* __restrict__ Wh,  const float* __restrict__ fcw,
    const float* __restrict__ fcb, float* __restrict__ out,
    float* __restrict__ hbuf, int* __restrict__ barrier_cnt, int nblocks)
{
  const int lane = threadIdx.x & 31;
  const int wid  = (blockIdx.x * blockDim.x + threadIdx.x) >> 5;   // 0..31
  const int mt   = wid & 1;          // batch-row tile (rows 16*mt..)
  const int jb   = wid >> 1;         // h-column tile  (cols 16*jb..)
  const int n    = lane & 15;
  const int kg   = lane >> 4;

  // Resident bf16 B fragments of W_h: 3 gates x 8 K-chunks = 192 VGPRs.
  // bf16 B(k,c): lane = n + 16*((k>>3)&1); vgpr v: k = 16*(v>>2)+8*kg+2*(v&3).
  BfVec Bf[3][8];
#pragma unroll
  for (int g = 0; g < 3; ++g) {
    const int c0 = g * H_ + jb * 16 + n;
#pragma unroll
    for (int ch = 0; ch < 8; ++ch) {
#pragma unroll
      for (int v = 0; v < 8; ++v) {
        const int k0 = ch * 32 + 16 * (v >> 2) + 8 * kg + 2 * (v & 3);
        Bf[g][ch].u[2 * v]     = f2bf(Wh[(size_t)k0 * G_ + c0]);
        Bf[g][ch].u[2 * v + 1] = f2bf(Wh[(size_t)(k0 + 1) * G_ + c0]);
      }
    }
  }

  float* h0 = hbuf;                 // [32][256], zero-initialized
  float* h1 = hbuf + B_ * H_;

  const int bbA = mt * 16 + n;      // A-fragment row (m = lane&15)
  const int mC  = 8 * kg;           // C row base (+v)
  const int jj  = jb * 16 + n;      // C column

  for (int t = 0; t < T_; ++t) {
    const float* hOld = (t & 1) ? h1 : h0;
    float*       hNew = (t & 1) ? h0 : h1;

    // Accumulators initialized from precomputed xg (states slots 1..3).
    v8f acc[3];
#pragma unroll
    for (int g = 0; g < 3; ++g) {
#pragma unroll
      for (int v = 0; v < 8; ++v) {
        const int bb = mt * 16 + mC + v;
        const size_t off =
            OUTBASE + ((((size_t)bb * 4 + (g + 1)) * T_ + t) * H_) + jj;
        acc[g][v] = out[off];
      }
    }

    // g += h_old @ W_h  (bf16 WMMA, K = 256 in 8 chunks of 32)
    const float* hr = hOld + (size_t)bbA * H_;
#pragma unroll
    for (int ch = 0; ch < 8; ++ch) {
      BfVec a;
#pragma unroll
      for (int v = 0; v < 8; ++v) {
        const int k0 = ch * 32 + 16 * (v >> 2) + 8 * kg + 2 * (v & 3);
        a.u[2 * v]     = f2bf(hr[k0]);
        a.u[2 * v + 1] = f2bf(hr[k0 + 1]);
      }
      acc[0] = __builtin_amdgcn_wmma_f32_16x16x32_bf16(
          false, a.v, false, Bf[0][ch].v, (short)0, acc[0], false, false);
      acc[1] = __builtin_amdgcn_wmma_f32_16x16x32_bf16(
          false, a.v, false, Bf[1][ch].v, (short)0, acc[1], false, false);
      acc[2] = __builtin_amdgcn_wmma_f32_16x16x32_bf16(
          false, a.v, false, Bf[2][ch].v, (short)0, acc[2], false, false);
    }

    // Gates + state writes (overwrite the xg slots we just consumed).
#pragma unroll
    for (int v = 0; v < 8; ++v) {
      const int bb = mt * 16 + mC + v;
      const float gr = acc[0][v], gz = acc[1][v], gn = acc[2][v];
      const float r  = sigmoidf_(gr);
      const float z  = sigmoidf_(gz);
      const float nn = tanhf(gn + r * gn);
      const float ho = hOld[(size_t)bb * H_ + jj];
      const float hn = (1.0f - z) * nn + z * ho;
      hNew[(size_t)bb * H_ + jj] = hn;
      const size_t sb = OUTBASE + (((size_t)bb * 4) * T_ + t) * H_ + jj;
      out[sb]                        = hn;   // s=0: h
      out[sb + (size_t)1 * T_ * H_]  = r;    // s=1: r
      out[sb + (size_t)2 * T_ * H_]  = z;    // s=2: z
      out[sb + (size_t)3 * T_ * H_]  = nn;   // s=3: n
    }

    // ---- grid-wide barrier (4 blocks, monotonic counter) ----
    __threadfence();                  // flush this thread's stores to L2
    __syncthreads();
    if (threadIdx.x == 0) {
      __hip_atomic_fetch_add(barrier_cnt, 1, __ATOMIC_RELEASE,
                             __HIP_MEMORY_SCOPE_AGENT);
      const int target = nblocks * (t + 1);
      while (__hip_atomic_load(barrier_cnt, __ATOMIC_ACQUIRE,
                               __HIP_MEMORY_SCOPE_AGENT) < target)
        __builtin_amdgcn_s_sleep(1);  // acquire load invalidates WGP$
    }
    __syncthreads();
  }

  // Final FC: output = h_last @ fc_w + fc_b  (h_last = h0, T even).
  {
    v8f acc = {};
    const float* hr = h0 + (size_t)bbA * H_;
    const int col = jb * 16 + n;
#pragma unroll 8
    for (int c = 0; c < 64; ++c) {
      const int k0 = c * 4 + 2 * kg;
      v2f a, b;
      a.x = hr[k0];                     a.y = hr[k0 + 1];
      b.x = fcw[(size_t)k0 * H_ + col]; b.y = fcw[(size_t)(k0 + 1) * H_ + col];
      acc = __builtin_amdgcn_wmma_f32_16x16x4_f32(
          false, a, false, b, (short)0, acc, false, false);
    }
    const float fb = fcb[col];
#pragma unroll
    for (int v = 0; v < 8; ++v) {
      const int bb = mt * 16 + mC + v;
      out[(size_t)bb * H_ + col] = acc[v] + fb;
    }
  }
}

// ---------------------------------------------------------------------------
extern "C" void kernel_launch(void* const* d_in, const int* in_sizes, int n_in,
                              void* d_out, int out_size, void* d_ws, size_t ws_size,
                              hipStream_t stream)
{
  (void)in_sizes; (void)n_in; (void)out_size; (void)ws_size;
  const float* x    = (const float*)d_in[0];
  const float* Wi   = (const float*)d_in[1];
  const float* Wh   = (const float*)d_in[2];
  const float* bias = (const float*)d_in[3];
  const float* fcw  = (const float*)d_in[4];
  const float* fcb  = (const float*)d_in[5];
  float* out = (float*)d_out;

  float* hbuf = (float*)d_ws;                                 // 2*32*256 f32
  int*   cnt  = (int*)((char*)d_ws + 2 * B_ * H_ * sizeof(float));

  hipMemsetAsync(d_ws, 0, 2 * B_ * H_ * sizeof(float) + sizeof(int), stream);

  // Phase 1: 24576 wave-jobs (4096 M-tiles/8 per wave x 48 N-tiles)
  xg_gemm_kernel<<<3072, 256, 0, stream>>>(x, Wi, bias, out);
  // Phase 2+3: 32 wave-jobs, 4 blocks x 256 threads
  gru_scan_kernel<<<4, 256, 0, stream>>>(Wh, fcw, fcb, out, hbuf, cnt, 4);
}
